// SketchRNN_26766236189018
// MI455X (gfx1250) — compile-verified, matrix operand order checked by reading, not run on hardware
//
#include <hip/hip_runtime.h>
#include <hip/hip_bf16.h>

// ---------------------------------------------------------------------------
// SketchRNN VAE forward for MI455X (gfx1250, wave32, WMMA).
// Strategy:
//   * hoist x@Wih^T projections for ALL timesteps into one big WMMA GEMM
//   * per-timestep fused kernel: 4-gate h@Whh^T WMMA + LSTM cell epilogue
//   * f16 activations/weights, f32 accumulation (v_wmma_f32_16x16x32_f16)
//   * weights converted to f16 once per call (~4.5MB -> L2 resident)
// ---------------------------------------------------------------------------

typedef __attribute__((ext_vector_type(16))) _Float16 v16h;
typedef __attribute__((ext_vector_type(8)))  float    v8f;

#define B_   512
#define T_   250
#define ENC_ 256
#define DEC_ 512
#define Z_   128
#define KDPAD 160   // decoder Wih K = 133 padded to 5*32
#define KEPAD 32    // encoder Wih K = 5 padded to 32
#define NMIXP 128   // mix head N = 123 padded to 128

__device__ __forceinline__ float sigmoidf_(float x) { return 1.0f / (1.0f + __expf(-x)); }

// 16-bit A/B fragment load, wave32 layout (16x32 tile, M (or N) = lane&15):
//   vgpr v holds K pair at  k = kbase + (v/4)*16 + (lane/16)*8 + (v%4)*2
__device__ __forceinline__ v16h load_tile_row(const _Float16* __restrict__ rowptr,
                                              int kbase, int half) {
  const unsigned* p = (const unsigned*)rowptr;
  union { v16h v; unsigned u[8]; } r;
#pragma unroll
  for (int i = 0; i < 8; ++i) {
    int k = kbase + ((i >> 2) << 4) + (half << 3) + ((i & 3) << 1);
    r.u[i] = p[k >> 1];
  }
  return r.v;
}

__device__ __forceinline__ v8f wmma_f16f32(v16h a, v16h b, v8f c) {
  // D = A(16x32,f16) x B(32x16,f16) + C(16x16,f32)
  return __builtin_amdgcn_wmma_f32_16x16x32_f16(
      /*neg_a=*/false, a, /*neg_b=*/false, b,
      /*c_mod=*/(short)0, c, /*reuse_a=*/false, /*reuse_b=*/false);
}

// ---------------------------------------------------------------------------
// Generic GEMM:  C[M][ldc] = A[M][lda](f16) @ W[N][ldw](f16)^T + bias
// One wave per 16(M) x 64(N) tile: 4 accumulators share one A fragment.
// ---------------------------------------------------------------------------
__global__ __launch_bounds__(256) void gemm4_kernel(
    const _Float16* __restrict__ A, int lda,
    const _Float16* __restrict__ W, int ldw,
    const float* __restrict__ bias,
    float* __restrict__ C, int ldc,
    int mTiles, int n4Tiles, int kChunks) {
  const int lane = threadIdx.x & 31;
  const int wave = blockIdx.x * (blockDim.x >> 5) + (threadIdx.x >> 5);
  if (wave >= mTiles * n4Tiles) return;
  const int mt = wave / n4Tiles, nt = wave % n4Tiles;
  const int m0 = mt * 16, n0 = nt * 64;
  const int half = lane >> 4, l = lane & 15;

  v8f acc[4];
#pragma unroll
  for (int j = 0; j < 4; ++j) {
    float bv = bias ? bias[n0 + j * 16 + l] : 0.0f;
#pragma unroll
    for (int i = 0; i < 8; ++i) acc[j][i] = bv;
  }
  const _Float16* arow = A + (size_t)(m0 + l) * lda;
  for (int kc = 0; kc < kChunks * 32; kc += 32) {
    v16h a = load_tile_row(arow, kc, half);
#pragma unroll
    for (int j = 0; j < 4; ++j) {
      v16h b = load_tile_row(W + (size_t)(n0 + j * 16 + l) * ldw, kc, half);
      acc[j] = wmma_f16f32(a, b, acc[j]);
    }
  }
#pragma unroll
  for (int j = 0; j < 4; ++j)
#pragma unroll
    for (int i = 0; i < 8; ++i)
      C[(size_t)(m0 + i + 8 * half) * ldc + n0 + j * 16 + l] = acc[j][i];
}

// ---------------------------------------------------------------------------
// Fused LSTM step: gates = xproj[t] + h_prev @ Whh^T ; cell update epilogue.
// One wave per 16(batch) x 16(hidden) tile; 4 accumulators = 4 gates.
// ---------------------------------------------------------------------------
template <int H, bool MASKED>
__global__ __launch_bounds__(256) void lstm_step_kernel(
    const float* __restrict__ xproj,       // [512][4H] slice for this t
    const _Float16* __restrict__ hprev,    // [512][H]
    const _Float16* __restrict__ W,        // [4H][H] f16 (Whh)
    _Float16* __restrict__ hnext,          // [512][H]
    float* __restrict__ c,                 // [512][H]
    const int* __restrict__ lengths, int t) {
  const int lane = threadIdx.x & 31;
  const int wave = blockIdx.x * (blockDim.x >> 5) + (threadIdx.x >> 5);
  const int ntiles = H / 16;
  if (wave >= (B_ / 16) * ntiles) return;
  const int mt = wave / ntiles, nt = wave % ntiles;
  const int m0 = mt * 16, n0 = nt * 16;
  const int half = lane >> 4, l = lane & 15;

  v8f acc[4];
#pragma unroll
  for (int g = 0; g < 4; ++g)
#pragma unroll
    for (int i = 0; i < 8; ++i)
      acc[g][i] = xproj[(size_t)(m0 + i + 8 * half) * (4 * H) + g * H + n0 + l];

  const _Float16* arow = hprev + (size_t)(m0 + l) * H;
  for (int kc = 0; kc < H; kc += 32) {
    v16h a = load_tile_row(arow, kc, half);
#pragma unroll
    for (int g = 0; g < 4; ++g) {
      v16h b = load_tile_row(W + (size_t)(g * H + n0 + l) * H, kc, half);
      acc[g] = wmma_f16f32(a, b, acc[g]);
    }
  }

#pragma unroll
  for (int i = 0; i < 8; ++i) {
    const int bi = m0 + i + 8 * half;
    const int n  = n0 + l;
    float co = c[(size_t)bi * H + n];
    float ig = sigmoidf_(acc[0][i]);
    float fg = sigmoidf_(acc[1][i]);
    float gg = tanhf(acc[2][i]);
    float og = sigmoidf_(acc[3][i]);
    float cn = fg * co + ig * gg;
    float hn = og * tanhf(cn);
    if constexpr (MASKED) {
      if (t >= lengths[bi]) { cn = co; hn = (float)hprev[(size_t)bi * H + n]; }
    }
    c[(size_t)bi * H + n] = cn;
    hnext[(size_t)bi * H + n] = (_Float16)hn;
  }
}

// ---------------------------------------------------------------------------
// Elementwise helpers
// ---------------------------------------------------------------------------
__global__ void cvtpad_kernel(const float* __restrict__ src, _Float16* __restrict__ dst,
                              int N, int Kk, int Npad, int Kpad) {
  int idx = blockIdx.x * blockDim.x + threadIdx.x;
  if (idx >= Npad * Kpad) return;
  int r = idx / Kpad, cc = idx % Kpad;
  float v = (r < N && cc < Kk) ? src[(size_t)r * Kk + cc] : 0.0f;
  dst[idx] = (_Float16)v;
}

__global__ void padbias_kernel(const float* __restrict__ src, float* __restrict__ dst,
                               int N, int Npad) {
  int idx = blockIdx.x * blockDim.x + threadIdx.x;
  if (idx < Npad) dst[idx] = (idx < N) ? src[idx] : 0.0f;
}

__global__ void zero16_kernel(_Float16* p, int n) {
  int idx = blockIdx.x * blockDim.x + threadIdx.x;
  if (idx < n) p[idx] = (_Float16)0.0f;
}
__global__ void zero32_kernel(float* p, int n) {
  int idx = blockIdx.x * blockDim.x + threadIdx.x;
  if (idx < n) p[idx] = 0.0f;
}

// encoder fwd input:  Af[t][b][32], cols 0..4 = data[b][t+1][:]
__global__ void pack_encf_kernel(const float* __restrict__ data, _Float16* __restrict__ dst) {
  int idx = blockIdx.x * blockDim.x + threadIdx.x;
  if (idx >= T_ * B_ * KEPAD) return;
  int t = idx / (B_ * KEPAD), rem = idx % (B_ * KEPAD);
  int b = rem / KEPAD, cc = rem % KEPAD;
  float v = (cc < 5) ? data[((size_t)b * (T_ + 1) + (t + 1)) * 5 + cc] : 0.0f;
  dst[idx] = (_Float16)v;
}

// encoder bwd input: gathered reverse  x_rev[b][t] = out_x[b][clip(len-1-t,0)]
__global__ void pack_encb_kernel(const float* __restrict__ data,
                                 const int* __restrict__ lengths,
                                 _Float16* __restrict__ dst) {
  int idx = blockIdx.x * blockDim.x + threadIdx.x;
  if (idx >= T_ * B_ * KEPAD) return;
  int t = idx / (B_ * KEPAD), rem = idx % (B_ * KEPAD);
  int b = rem / KEPAD, cc = rem % KEPAD;
  int ri = lengths[b] - 1 - t;
  if (ri < 0) ri = 0;
  float v = (cc < 5) ? data[((size_t)b * (T_ + 1) + (ri + 1)) * 5 + cc] : 0.0f;
  dst[idx] = (_Float16)v;
}

// decoder input x-part + zero pad: Ad[t][b][160], cols 0..4 = data[b][t][:]
__global__ void pack_decx_kernel(const float* __restrict__ data, _Float16* __restrict__ dst) {
  int idx = blockIdx.x * blockDim.x + threadIdx.x;
  if (idx >= T_ * B_ * KDPAD) return;
  int t = idx / (B_ * KDPAD), rem = idx % (B_ * KDPAD);
  int b = rem / KDPAD, cc = rem % KDPAD;
  float v = (cc < 5) ? data[((size_t)b * (T_ + 1) + t) * 5 + cc] : 0.0f;
  dst[idx] = (_Float16)v;
}

// z = mean + exp(0.5*logvar) * eps   (zstats = [B][256], mean|logvar)
__global__ void z_kernel(const float* __restrict__ zstats, const float* __restrict__ eps,
                         _Float16* __restrict__ z16) {
  int idx = blockIdx.x * blockDim.x + threadIdx.x;
  if (idx >= B_ * Z_) return;
  int b = idx / Z_, j = idx % Z_;
  float z = zstats[b * 2 * Z_ + j] + __expf(0.5f * zstats[b * 2 * Z_ + Z_ + j]) * eps[idx];
  z16[idx] = (_Float16)z;
}

// broadcast z into decoder input cols 5..132 for every timestep
__global__ void zbroadcast_kernel(const _Float16* __restrict__ z16, _Float16* __restrict__ decA) {
  int idx = blockIdx.x * blockDim.x + threadIdx.x;
  if (idx >= T_ * B_ * Z_) return;
  int t = idx / (B_ * Z_), rem = idx % (B_ * Z_);
  int b = rem / Z_, j = rem % Z_;
  decA[((size_t)t * B_ + b) * KDPAD + 5 + j] = z16[b * Z_ + j];
}

// h0,c0 = split(tanh(init)) ; h0 -> hs slot 0 (f16), c0 -> c_dec (f32)
__global__ void init_split_kernel(const float* __restrict__ initraw,
                                  _Float16* __restrict__ hs0, float* __restrict__ cdec) {
  int idx = blockIdx.x * blockDim.x + threadIdx.x;
  if (idx >= B_ * DEC_) return;
  int b = idx / DEC_, j = idx % DEC_;
  hs0[idx]  = (_Float16)tanhf(initraw[(size_t)b * 2 * DEC_ + j]);
  cdec[idx] = tanhf(initraw[(size_t)b * 2 * DEC_ + DEC_ + j]);
}

__global__ void feat_copy_kernel(const _Float16* __restrict__ h, _Float16* __restrict__ feat,
                                 int off) {
  int idx = blockIdx.x * blockDim.x + threadIdx.x;
  if (idx >= B_ * ENC_) return;
  int b = idx / ENC_, j = idx % ENC_;
  feat[(size_t)b * (2 * ENC_) + off + j] = h[idx];
}

// MDN head: pi softmax(20) | mu_x | mu_y | exp sigma_x | exp sigma_y | tanh rho | q softmax(3)
__global__ void mix_epilogue_kernel(const float* __restrict__ y, float* __restrict__ out) {
  int r = blockIdx.x * blockDim.x + threadIdx.x;
  if (r >= T_ * B_) return;
  int t = r >> 9, b = r & (B_ - 1);
  const float* yr = y + (size_t)r * NMIXP;
  float* o = out + ((size_t)b * T_ + t) * 123;
  float mx = yr[0];
  for (int i = 1; i < 20; ++i) mx = fmaxf(mx, yr[i]);
  float s = 0.0f, e[20];
  for (int i = 0; i < 20; ++i) { e[i] = __expf(yr[i] - mx); s += e[i]; }
  float inv = 1.0f / s;
  for (int i = 0; i < 20; ++i) o[i] = e[i] * inv;
  for (int i = 20; i < 60; ++i) o[i] = yr[i];
  for (int i = 60; i < 100; ++i) o[i] = __expf(yr[i]);
  for (int i = 100; i < 120; ++i) o[i] = tanhf(yr[i]);
  float m2 = fmaxf(yr[120], fmaxf(yr[121], yr[122]));
  float q0 = __expf(yr[120] - m2), q1 = __expf(yr[121] - m2), q2 = __expf(yr[122] - m2);
  float qi = 1.0f / (q0 + q1 + q2);
  o[120] = q0 * qi; o[121] = q1 * qi; o[122] = q2 * qi;
}

// ---------------------------------------------------------------------------
extern "C" void kernel_launch(void* const* d_in, const int* in_sizes, int n_in,
                              void* d_out, int out_size, void* d_ws, size_t ws_size,
                              hipStream_t stream) {
  (void)in_sizes; (void)n_in; (void)out_size; (void)ws_size;
  const float* data      = (const float*)d_in[0];
  const int*   lengths   = (const int*)d_in[1];
  const float* eps       = (const float*)d_in[2];
  const float* enc_Wih_f = (const float*)d_in[3];
  const float* enc_Whh_f = (const float*)d_in[4];
  const float* enc_b_f   = (const float*)d_in[5];
  const float* enc_Wih_b = (const float*)d_in[6];
  const float* enc_Whh_b = (const float*)d_in[7];
  const float* enc_b_b   = (const float*)d_in[8];
  const float* enc_out_W = (const float*)d_in[9];
  const float* enc_out_b = (const float*)d_in[10];
  const float* init_W    = (const float*)d_in[11];
  const float* init_b    = (const float*)d_in[12];
  const float* dec_Wih   = (const float*)d_in[13];
  const float* dec_Whh   = (const float*)d_in[14];
  const float* dec_b     = (const float*)d_in[15];
  const float* mix_W     = (const float*)d_in[16];
  const float* mix_b     = (const float*)d_in[17];
  float* out = (float*)d_out;

  // ---- workspace carve-up (256B aligned) ----
  char* base = (char*)d_ws;
  size_t off = 0;
  auto alloc = [&](size_t bytes) {
    size_t o = off;
    off = (off + bytes + 255) & ~(size_t)255;
    return base + o;
  };
  // big region reused: encF xproj -> encB xproj -> dec xproj
  float*    xproj  = (float*)alloc((size_t)T_ * B_ * 4 * DEC_ * 4);          // 1.05 GB
  _Float16* Abuf   = (_Float16*)alloc((size_t)T_ * B_ * KDPAD * 2);         // 41 MB (reused)
  _Float16* hsbuf  = (_Float16*)alloc((size_t)(T_ + 1) * B_ * DEC_ * 2);    // 132 MB
  float*    yraw   = (float*)alloc((size_t)T_ * B_ * NMIXP * 4);            // 65.5 MB
  _Float16* wihF   = (_Float16*)alloc((size_t)4 * ENC_ * KEPAD * 2);
  _Float16* whhF   = (_Float16*)alloc((size_t)4 * ENC_ * ENC_ * 2);
  _Float16* wihB   = (_Float16*)alloc((size_t)4 * ENC_ * KEPAD * 2);
  _Float16* whhB   = (_Float16*)alloc((size_t)4 * ENC_ * ENC_ * 2);
  _Float16* dwih   = (_Float16*)alloc((size_t)4 * DEC_ * KDPAD * 2);
  _Float16* dwhh   = (_Float16*)alloc((size_t)4 * DEC_ * DEC_ * 2);
  _Float16* eow16  = (_Float16*)alloc((size_t)2 * Z_ * 2 * ENC_ * 2);
  _Float16* iw16   = (_Float16*)alloc((size_t)2 * DEC_ * Z_ * 2);
  _Float16* mw16   = (_Float16*)alloc((size_t)NMIXP * DEC_ * 2);
  float*    mixbp  = (float*)alloc(NMIXP * 4);
  _Float16* h0b    = (_Float16*)alloc((size_t)B_ * ENC_ * 2);
  _Float16* h1b    = (_Float16*)alloc((size_t)B_ * ENC_ * 2);
  float*    c_enc  = (float*)alloc((size_t)B_ * ENC_ * 4);
  _Float16* feat   = (_Float16*)alloc((size_t)B_ * 2 * ENC_ * 2);
  float*    zstats = (float*)alloc((size_t)B_ * 2 * Z_ * 4);
  _Float16* z16    = (_Float16*)alloc((size_t)B_ * Z_ * 2);
  float*    initrw = (float*)alloc((size_t)B_ * 2 * DEC_ * 4);
  float*    c_dec  = (float*)alloc((size_t)B_ * DEC_ * 4);

  auto gb = [](size_t n) { return (unsigned)((n + 255) / 256); };

  // ---- pack weights to f16 (L2-resident thereafter) ----
  cvtpad_kernel<<<gb(1024 * KEPAD), 256, 0, stream>>>(enc_Wih_f, wihF, 1024, 5, 1024, KEPAD);
  cvtpad_kernel<<<gb(1024 * 256), 256, 0, stream>>>(enc_Whh_f, whhF, 1024, 256, 1024, 256);
  cvtpad_kernel<<<gb(1024 * KEPAD), 256, 0, stream>>>(enc_Wih_b, wihB, 1024, 5, 1024, KEPAD);
  cvtpad_kernel<<<gb(1024 * 256), 256, 0, stream>>>(enc_Whh_b, whhB, 1024, 256, 1024, 256);
  cvtpad_kernel<<<gb(2048 * KDPAD), 256, 0, stream>>>(dec_Wih, dwih, 2048, 133, 2048, KDPAD);
  cvtpad_kernel<<<gb(2048 * 512), 256, 0, stream>>>(dec_Whh, dwhh, 2048, 512, 2048, 512);
  cvtpad_kernel<<<gb(256 * 512), 256, 0, stream>>>(enc_out_W, eow16, 256, 512, 256, 512);
  cvtpad_kernel<<<gb(1024 * 128), 256, 0, stream>>>(init_W, iw16, 1024, 128, 1024, 128);
  cvtpad_kernel<<<gb(NMIXP * 512), 256, 0, stream>>>(mix_W, mw16, 123, 512, NMIXP, 512);
  padbias_kernel<<<1, 128, 0, stream>>>(mix_b, mixbp, 123, NMIXP);

  const int mT = T_ * B_ / 16;  // 8000 row tiles for the batched GEMMs

  // ---- encoder forward ----
  pack_encf_kernel<<<gb((size_t)T_ * B_ * KEPAD), 256, 0, stream>>>(data, Abuf);
  gemm4_kernel<<<(mT * 16) / 8, 256, 0, stream>>>(Abuf, KEPAD, wihF, KEPAD, enc_b_f,
                                                  xproj, 4 * ENC_, mT, 16, 1);
  zero16_kernel<<<gb(B_ * ENC_), 256, 0, stream>>>(h0b, B_ * ENC_);
  zero16_kernel<<<gb(B_ * ENC_), 256, 0, stream>>>(h1b, B_ * ENC_);
  zero32_kernel<<<gb(B_ * ENC_), 256, 0, stream>>>(c_enc, B_ * ENC_);
  {
    _Float16* ha = h0b; _Float16* hbuf = h1b;
    for (int t = 0; t < T_; ++t) {
      lstm_step_kernel<ENC_, true><<<64, 256, 0, stream>>>(
          xproj + (size_t)t * B_ * 4 * ENC_, ha, whhF, hbuf, c_enc, lengths, t);
      _Float16* tmp = ha; ha = hbuf; hbuf = tmp;
    }
    feat_copy_kernel<<<gb(B_ * ENC_), 256, 0, stream>>>(ha, feat, 0);
  }

  // ---- encoder backward (gathered reverse sequence, same mask) ----
  pack_encb_kernel<<<gb((size_t)T_ * B_ * KEPAD), 256, 0, stream>>>(data, lengths, Abuf);
  gemm4_kernel<<<(mT * 16) / 8, 256, 0, stream>>>(Abuf, KEPAD, wihB, KEPAD, enc_b_b,
                                                  xproj, 4 * ENC_, mT, 16, 1);
  zero16_kernel<<<gb(B_ * ENC_), 256, 0, stream>>>(h0b, B_ * ENC_);
  zero16_kernel<<<gb(B_ * ENC_), 256, 0, stream>>>(h1b, B_ * ENC_);
  zero32_kernel<<<gb(B_ * ENC_), 256, 0, stream>>>(c_enc, B_ * ENC_);
  {
    _Float16* ha = h0b; _Float16* hbuf = h1b;
    for (int t = 0; t < T_; ++t) {
      lstm_step_kernel<ENC_, true><<<64, 256, 0, stream>>>(
          xproj + (size_t)t * B_ * 4 * ENC_, ha, whhB, hbuf, c_enc, lengths, t);
      _Float16* tmp = ha; ha = hbuf; hbuf = tmp;
    }
    feat_copy_kernel<<<gb(B_ * ENC_), 256, 0, stream>>>(ha, feat, ENC_);
  }

  // ---- latent: enc_out GEMM, reparameterize, init GEMM ----
  gemm4_kernel<<<(32 * 4) / 8, 256, 0, stream>>>(feat, 512, eow16, 512, enc_out_b,
                                                 zstats, 256, 32, 4, 16);
  z_kernel<<<gb(B_ * Z_), 256, 0, stream>>>(zstats, eps, z16);
  pack_decx_kernel<<<gb((size_t)T_ * B_ * KDPAD), 256, 0, stream>>>(data, Abuf);
  zbroadcast_kernel<<<gb((size_t)T_ * B_ * Z_), 256, 0, stream>>>(z16, Abuf);
  gemm4_kernel<<<(32 * 16) / 8, 256, 0, stream>>>(z16, 128, iw16, 128, init_b,
                                                  initrw, 1024, 32, 16, 4);
  init_split_kernel<<<gb(B_ * DEC_), 256, 0, stream>>>(initrw, hsbuf, c_dec);

  // ---- decoder: batched input projection + 250 fused recurrent steps ----
  gemm4_kernel<<<(mT * 32) / 8, 256, 0, stream>>>(Abuf, KDPAD, dwih, KDPAD, dec_b,
                                                  xproj, 4 * DEC_, mT, 32, 5);
  for (int t = 0; t < T_; ++t) {
    lstm_step_kernel<DEC_, false><<<128, 256, 0, stream>>>(
        xproj + (size_t)t * B_ * 4 * DEC_,
        hsbuf + (size_t)t * B_ * DEC_, dwhh,
        hsbuf + (size_t)(t + 1) * B_ * DEC_, c_dec, nullptr, t);
  }

  // ---- mixture head GEMM + MDN epilogue ----
  gemm4_kernel<<<(mT * 2) / 8, 256, 0, stream>>>(hsbuf + (size_t)B_ * DEC_, DEC_,
                                                 mw16, DEC_, mixbp,
                                                 yraw, NMIXP, mT, 2, 16);
  mix_epilogue_kernel<<<gb(T_ * B_), 256, 0, stream>>>(yraw, out);
}
